// MultiHeadAttention_45311904973389
// MI455X (gfx1250) — matrix operand order repaired and checked by reading, not compiled, not run on hardware
//
#include <hip/hip_runtime.h>

// Problem constants (reference: B=2, S=2048, D=1024, H=16, DK=64)
#define BB   2
#define SS   2048
#define DD   1024
#define HH   16
#define DKK  64

typedef __bf16  v16bf __attribute__((ext_vector_type(16)));
typedef __bf16  v8bf  __attribute__((ext_vector_type(8)));
typedef __bf16  v4bf  __attribute__((ext_vector_type(4)));
typedef float   v8f   __attribute__((ext_vector_type(8)));
typedef float   v4f   __attribute__((ext_vector_type(4)));

// ---------------------------------------------------------------- helpers ---

__device__ __forceinline__ __bf16 f2bf(float f) {
  unsigned u = __builtin_bit_cast(unsigned, f);
  unsigned r = (u + 0x7FFFu + ((u >> 16) & 1u)) >> 16;   // round-to-nearest-even
  unsigned short s = (unsigned short)r;
  return __builtin_bit_cast(__bf16, s);
}

__device__ __forceinline__ v8f wmma_bf16(v16bf a, v16bf b, v8f c) {
  return __builtin_amdgcn_wmma_f32_16x16x32_bf16(
      /*neg_a=*/false, a, /*neg_b=*/false, b,
      /*c_mod=*/(short)0, c, /*reuse_a=*/false, /*reuse_b=*/false);
}

// Async copy 16B global -> LDS (CDNA5 GLOBAL_LOAD_ASYNC_TO_LDS_B128, ASYNCcnt).
// LDS dest address = low 32 bits of the flat shared pointer.
__device__ __forceinline__ void async_ld16(const void* g, void* l) {
  unsigned loff = (unsigned)(unsigned long long)(uintptr_t)l;
  unsigned long long ga = (unsigned long long)(uintptr_t)g;
  asm volatile("global_load_async_to_lds_b128 %0, %1, off"
               :: "v"(loff), "v"(ga) : "memory");
}
__device__ __forceinline__ void wait_async0() {
  asm volatile("s_wait_asynccnt 0x0" ::: "memory");
}

// A-operand (16x32 bf16): lane L -> row L&15; elems 0..7 = K base0..base0+7,
// elems 8..15 = K 16+base0 .. 16+base0+7, base0 = (L>=16)?8:0.
__device__ __forceinline__ v16bf load_a_bf16(const __bf16* rowp, int base0) {
  v8bf lo = *(const v8bf*)(rowp + base0);
  v8bf hi = *(const v8bf*)(rowp + 16 + base0);
  v16bf a;
#pragma unroll
  for (int i = 0; i < 8; ++i) { a[i] = lo[i]; a[8 + i] = hi[i]; }
  return a;
}

// ------------------------------------- kernel 0a: f32 -> bf16 streaming ----
__global__ __launch_bounds__(256) void convert_bf16(
    const float* __restrict__ x, __bf16* __restrict__ y, int n) {
  int i = (blockIdx.x * 256 + threadIdx.x) * 4;
  if (i < n) {
    v4f f = *(const v4f*)(x + i);
    v4bf o;
#pragma unroll
    for (int j = 0; j < 4; ++j) o[j] = f2bf(f[j]);
    *(v4bf*)(y + i) = o;
  }
}

// ------------------------------------- kernel 0b: W -> Wt (bf16, transposed)
// Wt[k*D + n] = (bf16) W[n*D + k]
__global__ __launch_bounds__(256) void transpose_convert(
    const float* __restrict__ W, __bf16* __restrict__ Wt) {
  __shared__ float tile[32][33];
  int bx = blockIdx.x * 32, by = blockIdx.y * 32;
  int tx = threadIdx.x, ty = threadIdx.y;       // block (32, 8)
#pragma unroll
  for (int i = ty; i < 32; i += 8)
    tile[i][tx] = W[(size_t)(by + i) * DD + bx + tx];
  __syncthreads();
#pragma unroll
  for (int i = ty; i < 32; i += 8)
    Wt[(size_t)(bx + i) * DD + by + tx] = f2bf(tile[tx][i]);
}

// ------------------------------- kernel 1/4: GEMM  Y = X @ W^T + bias -------
// M=4096 (B*S), N=K=1024.  Block = 128 thr = 4 waves; wave -> 32x64 tile,
// block -> 128x64.  B (weight) tile 32x64 staged via async global->LDS,
// double-buffered, shared by all 4 waves.  A prefetched one k-step ahead.
// B fragments batch-loaded from LDS before the WMMA chain (staggered waits).
// mode 0: Q head-major  [B,H,S,DK] bf16 | mode 1: K^T [B,H,DK,S] bf16
// mode 2: V head-major  [B,H,S,DK] bf16 | mode 3: f32 row-major [M,D]
__global__ __launch_bounds__(128) void proj_gemm(
    const __bf16* __restrict__ Xb, const __bf16* __restrict__ Wt,
    const float* __restrict__ bias,
    __bf16* __restrict__ outB, float* __restrict__ outF, int mode) {
  __shared__ __align__(128) __bf16 btile[2][32][64];   // 2 x 4KB

  int tid   = threadIdx.x;
  int lane  = tid & 31;
  int w     = tid >> 5;
  int m0    = blockIdx.y * 128 + w * 32;
  int n0    = blockIdx.x * 64;
  int row16 = lane & 15;
  int base0 = (lane >> 4) << 3;

  const __bf16* xrow0 = Xb + (size_t)(m0 + row16) * DD;
  const __bf16* xrow1 = Xb + (size_t)(m0 + 16 + row16) * DD;

  // prologue: stage B(kk=0), load A(kk=0)
#pragma unroll
  for (int i = 0; i < 2; ++i) {
    int si = i * 128 + tid, row = si >> 3, sg = si & 7;
    async_ld16(Wt + (size_t)row * DD + n0 + sg * 8, &btile[0][row][sg * 8]);
  }
  v16bf a0 = load_a_bf16(xrow0, base0);
  v16bf a1 = load_a_bf16(xrow1, base0);

  v8f acc[2][4] = {};

#pragma unroll 2
  for (int kk = 0; kk < DD; kk += 32) {
    int buf = (kk >> 5) & 1;
    wait_async0();            // staged B tile for this kk is in LDS
    __syncthreads();          // ...for every wave; prev buffer fully consumed
    bool more = (kk + 32 < DD);
    if (more) {               // stage next B tile while we compute this one
#pragma unroll
      for (int i = 0; i < 2; ++i) {
        int si = i * 128 + tid, row = si >> 3, sg = si & 7;
        async_ld16(Wt + (size_t)(kk + 32 + row) * DD + n0 + sg * 8,
                   &btile[buf ^ 1][row][sg * 8]);
      }
    }
    v16bf na0 = a0, na1 = a1;
    if (more) {
      na0 = load_a_bf16(xrow0 + kk + 32, base0);
      na1 = load_a_bf16(xrow1 + kk + 32, base0);
    }
    // batch all B fragments, then run the WMMA chain
    v16bf bfr[4];
#pragma unroll
    for (int t = 0; t < 4; ++t)
      for (int s = 0; s < 16; ++s)
        bfr[t][s] = btile[buf][base0 + (s & 7) + ((s >> 3) << 4)][16 * t + row16];      // elem s = K row (table order), lane = N col
#pragma unroll
    for (int t = 0; t < 4; ++t) {
      acc[0][t] = wmma_bf16(a0, bfr[t], acc[0][t]);
      acc[1][t] = wmma_bf16(a1, bfr[t], acc[1][t]);
    }
    a0 = na0; a1 = na1;
  }

#pragma unroll
  for (int mi = 0; mi < 2; ++mi) {
#pragma unroll
    for (int t = 0; t < 4; ++t) {
      int   ncol = n0 + 16 * t + (lane & 15);
      float bvv  = bias[ncol];
#pragma unroll
      for (int r = 0; r < 8; ++r) {
        int   m = m0 + mi * 16 + r + ((lane >> 4) << 3);
        float y = acc[mi][t][r] + bvv;
        if (mode == 3) {
          outF[(size_t)m * DD + ncol] = y;
        } else {
          int b2 = m >> 11, s = m & (SS - 1);
          int h  = ncol >> 6, dk = ncol & (DKK - 1);
          if (mode == 1)
            outB[((size_t)(b2 * HH + h) * DKK + dk) * SS + s] = f2bf(y);  // K^T
          else
            outB[((size_t)(b2 * HH + h) * SS + s) * DKK + dk] = f2bf(y);  // Q/V
        }
      }
    }
  }
}

// ------------------------------- kernel 2: flash attention per (b,h) --------
// Q [S,DK] bf16, Kt [DK,S] bf16, V [S,DK] bf16 -> ctx bf16 [B*S, D]
// grid (S/64, B*H), block 128 (4 waves); wave owns 16 q-rows, streams 64 keys
// per iteration.  K tile (64x64) and V tile (64x64) staged via async
// global->LDS, double-buffered, shared by all 4 waves of the block.
__global__ __launch_bounds__(128) void attention_kernel(
    const __bf16* __restrict__ Qh, const __bf16* __restrict__ Ktr,
    const __bf16* __restrict__ Vh, __bf16* __restrict__ ctx) {
  __shared__ __align__(128) __bf16 kt[2][64][64];    // 2 x 8KB  (rows = d)
  __shared__ __align__(128) __bf16 vt[2][64][64];    // 2 x 8KB  (rows = key)
  __shared__ __align__(128) __bf16 pbuf[4][16][64];  // per-wave P tile, 8KB

  int tid   = threadIdx.x;
  int lane  = tid & 31;
  int w     = tid >> 5;
  int bh    = blockIdx.y;
  int b2    = bh >> 4, h = bh & 15;
  int m0    = blockIdx.x * 64 + w * 16;
  int row16 = lane & 15;
  int base0 = (lane >> 4) << 3;

  const __bf16* Q = Qh  + (size_t)bh * SS * DKK;
  const __bf16* K = Ktr + (size_t)bh * DKK * SS;
  const __bf16* V = Vh  + (size_t)bh * SS * DKK;

  // Q A-fragments stay in registers for the whole sweep: K 0..31 and 32..63
  const __bf16* qrow = Q + (size_t)(m0 + row16) * DKK;
  v16bf aq0 = load_a_bf16(qrow, base0);
  v16bf aq1 = load_a_bf16(qrow + 32, base0);

  // prologue stage of K/V tiles for j0 = 0 (8KB each: 512 x 16B segments)
#pragma unroll
  for (int i = 0; i < 4; ++i) {
    int si = i * 128 + tid, row = si >> 3, sg = si & 7;
    async_ld16(K + (size_t)row * SS + sg * 8, &kt[0][row][sg * 8]);
  }
#pragma unroll
  for (int i = 0; i < 4; ++i) {
    int si = i * 128 + tid, row = si >> 3, sg = si & 7;
    async_ld16(V + (size_t)row * DKK + sg * 8, &vt[0][row][sg * 8]);
  }

  v8f   acc[4] = {};
  float mrow[8], lrow[8];
#pragma unroll
  for (int r = 0; r < 8; ++r) { mrow[r] = -1e30f; lrow[r] = 0.0f; }

  const float scale = 0.125f;   // 1/sqrt(64)

#pragma unroll 2
  for (int j0 = 0; j0 < SS; j0 += 64) {
    int buf = (j0 >> 6) & 1;
    wait_async0();
    __syncthreads();
    if (j0 + 64 < SS) {   // stage next K/V tiles while computing on this one
#pragma unroll
      for (int i = 0; i < 4; ++i) {
        int si = i * 128 + tid, row = si >> 3, sg = si & 7;
        async_ld16(K + (size_t)row * SS + j0 + 64 + sg * 8,
                   &kt[buf ^ 1][row][sg * 8]);
      }
#pragma unroll
      for (int i = 0; i < 4; ++i) {
        int si = i * 128 + tid, row = si >> 3, sg = si & 7;
        async_ld16(V + (size_t)(j0 + 64 + row) * DKK + sg * 8,
                   &vt[buf ^ 1][row][sg * 8]);
      }
    }

    // ---- scores: 16 q-rows x 64 keys, f32 acc (batched B frags from LDS) --
    v8f sc[4];
    {
      v16bf bk[4];
#pragma unroll
      for (int t = 0; t < 4; ++t)
        for (int s = 0; s < 16; ++s)
          bk[t][s] = kt[buf][base0 + (s & 7) + ((s >> 3) << 4)][16 * t + row16];          // d = elem s (table order), key = lane
#pragma unroll
      for (int t = 0; t < 4; ++t) {
        v8f z = {};
        sc[t] = wmma_bf16(aq0, bk[t], z);
      }
#pragma unroll
      for (int t = 0; t < 4; ++t)
        for (int s = 0; s < 16; ++s)
          bk[t][s] = kt[buf][32 + base0 + (s & 7) + ((s >> 3) << 4)][16 * t + row16];     // d = 32+elem s (table order)
#pragma unroll
      for (int t = 0; t < 4; ++t) {
        sc[t] = wmma_bf16(aq1, bk[t], sc[t]);
#pragma unroll
        for (int r = 0; r < 8; ++r) sc[t][r] *= scale;
      }
    }

    // ---- online softmax (row stats replicated across each 16-lane half) --
#pragma unroll
    for (int r = 0; r < 8; ++r) {
      float mx = fmaxf(fmaxf(sc[0][r], sc[1][r]), fmaxf(sc[2][r], sc[3][r]));
#pragma unroll
      for (int msk = 1; msk < 16; msk <<= 1)
        mx = fmaxf(mx, __shfl_xor(mx, msk, 32));
      float mnew = fmaxf(mrow[r], mx);
      float corr = __expf(mrow[r] - mnew);
      float p[4], ps = 0.0f;
#pragma unroll
      for (int t = 0; t < 4; ++t) { p[t] = __expf(sc[t][r] - mnew); ps += p[t]; }
#pragma unroll
      for (int msk = 1; msk < 16; msk <<= 1)
        ps += __shfl_xor(ps, msk, 32);
      lrow[r] = lrow[r] * corr + ps;
      mrow[r] = mnew;
#pragma unroll
      for (int t = 0; t < 4; ++t) acc[t][r] *= corr;
      int rr = r + ((lane >> 4) << 3);
#pragma unroll
      for (int t = 0; t < 4; ++t)
        pbuf[w][rr][16 * t + (lane & 15)] = f2bf(p[t]);   // C-layout -> LDS
    }

    // ---- P (16x64) back as two A-operands; accumulate against V (LDS) ----
    v16bf ap0 = load_a_bf16(&pbuf[w][row16][0], base0);    // keys 0..31
    v16bf ap1 = load_a_bf16(&pbuf[w][row16][32], base0);   // keys 32..63
    {
      v16bf bv[4];
#pragma unroll
      for (int t = 0; t < 4; ++t)
        for (int s = 0; s < 16; ++s)
          bv[t][s] = vt[buf][base0 + (s & 7) + ((s >> 3) << 4)][16 * t + row16];          // key = elem s (table order), d = lane
#pragma unroll
      for (int t = 0; t < 4; ++t)
        acc[t] = wmma_bf16(ap0, bv[t], acc[t]);
#pragma unroll
      for (int t = 0; t < 4; ++t)
        for (int s = 0; s < 16; ++s)
          bv[t][s] = vt[buf][32 + base0 + (s & 7) + ((s >> 3) << 4)][16 * t + row16];     // key = 32+elem s (table order)
#pragma unroll
      for (int t = 0; t < 4; ++t)
        acc[t] = wmma_bf16(ap1, bv[t], acc[t]);
    }
  }

  // ---- normalize and scatter into concat layout [B*S, D] -----------------
  int colb = h * DKK;
  int rowb = b2 * SS + m0;
#pragma unroll
  for (int r = 0; r < 8; ++r) {
    float inv = 1.0f / lrow[r];
    int   row = r + ((lane >> 4) << 3);
    size_t off = (size_t)(rowb + row) * DD + colb + (lane & 15);
#pragma unroll
    for (int t = 0; t < 4; ++t)
      ctx[off + 16 * t] = f2bf(acc[t][r] * inv);
  }
}

// ---------------------------------------------------------------- launch ----

extern "C" void kernel_launch(void* const* d_in, const int* in_sizes, int n_in,
                              void* d_out, int out_size, void* d_ws, size_t ws_size,
                              hipStream_t stream) {
  const float* q  = (const float*)d_in[0];
  const float* k  = (const float*)d_in[1];
  const float* v  = (const float*)d_in[2];
  const float* Wq = (const float*)d_in[3];
  const float* bq = (const float*)d_in[4];
  const float* Wk = (const float*)d_in[5];
  const float* bk = (const float*)d_in[6];
  const float* Wv = (const float*)d_in[7];
  const float* bv = (const float*)d_in[8];
  const float* Wo = (const float*)d_in[9];
  const float* bo = (const float*)d_in[10];
  float* out = (float*)d_out;

  char* ws = (char*)d_ws;
  const size_t MB = 1u << 20;
  __bf16* Wqt = (__bf16*)(ws + 0 * MB);    // 2 MB each, bf16 transposed weights
  __bf16* Wkt = (__bf16*)(ws + 2 * MB);
  __bf16* Wvt = (__bf16*)(ws + 4 * MB);
  __bf16* Wot = (__bf16*)(ws + 6 * MB);
  __bf16* qb  = (__bf16*)(ws + 8 * MB);    // 8 MB each, bf16 activations
  __bf16* kb  = (__bf16*)(ws + 16 * MB);
  __bf16* vb  = (__bf16*)(ws + 24 * MB);
  __bf16* Qh  = (__bf16*)(ws + 32 * MB);   // 8 MB  [B,H,S,DK]
  __bf16* Ktr = (__bf16*)(ws + 40 * MB);   // 8 MB  [B,H,DK,S]
  __bf16* Vh  = (__bf16*)(ws + 48 * MB);   // 8 MB  [B,H,S,DK]
  __bf16* ctx = (__bf16*)(ws + 56 * MB);   // 8 MB  [B*S, D]

  const int NACT = BB * SS * DD;           // 4M elements
  convert_bf16<<<NACT / (256 * 4), 256, 0, stream>>>(q, qb, NACT);
  convert_bf16<<<NACT / (256 * 4), 256, 0, stream>>>(k, kb, NACT);
  convert_bf16<<<NACT / (256 * 4), 256, 0, stream>>>(v, vb, NACT);

  dim3 tb(32, 8), tg(DD / 32, DD / 32);
  transpose_convert<<<tg, tb, 0, stream>>>(Wq, Wqt);
  transpose_convert<<<tg, tb, 0, stream>>>(Wk, Wkt);
  transpose_convert<<<tg, tb, 0, stream>>>(Wv, Wvt);
  transpose_convert<<<tg, tb, 0, stream>>>(Wo, Wot);

  dim3 gb(128);
  dim3 gg(DD / 64, (BB * SS) / 128);
  proj_gemm<<<gg, gb, 0, stream>>>(qb,  Wqt, bq, Qh,  nullptr, 0);
  proj_gemm<<<gg, gb, 0, stream>>>(kb,  Wkt, bk, Ktr, nullptr, 1);
  proj_gemm<<<gg, gb, 0, stream>>>(vb,  Wvt, bv, Vh,  nullptr, 2);

  attention_kernel<<<dim3(SS / 64, BB * HH), gb, 0, stream>>>(Qh, Ktr, Vh, ctx);

  proj_gemm<<<gg, gb, 0, stream>>>(ctx, Wot, bo, nullptr, out, 3);
}
